// ContextAwareRouter_83897891160586
// MI455X (gfx1250) — compile-verified, matrix operand order checked by reading, not run on hardware
//
#include <hip/hip_runtime.h>
#include <hip/hip_bf16.h>

typedef __attribute__((ext_vector_type(16))) __bf16 v16bf;
typedef __attribute__((ext_vector_type(8)))  __bf16 v8bf;
typedef __attribute__((ext_vector_type(4)))  __bf16 v4bf;
typedef __attribute__((ext_vector_type(8)))  float  v8f;
typedef __attribute__((ext_vector_type(4)))  int    v4i;

union Frag16 { v16bf v; v8bf h[2]; };

static __device__ __forceinline__ v8f wmma_bf16(const Frag16& a, const Frag16& b, v8f c) {
    return __builtin_amdgcn_wmma_f32_16x16x32_bf16(false, a.v, false, b.v,
                                                   (short)0, c, false, false);
}

// ---------------------------------------------------------------------------
// CDNA5 async global->LDS copy (16 bytes per lane), ASYNCcnt-tracked.
// Toolchain signature (from clang diagnostic): (AS1 v4i*, AS3 v4i*, Ii, Ii).
// Falls back to a plain LDS store if the builtin is unavailable.
// ---------------------------------------------------------------------------
static __device__ __forceinline__ void async_copy_b128(const void* g, void* l) {
#if defined(__gfx1250__) && __has_builtin(__builtin_amdgcn_global_load_async_to_lds_b128)
    __builtin_amdgcn_global_load_async_to_lds_b128(
        (__attribute__((address_space(1))) v4i*)g,
        (__attribute__((address_space(3))) v4i*)l,
        /*offset=*/0, /*cpol=*/0);
#else
    *(v8bf*)l = *(const v8bf*)g;
#endif
}

static __device__ __forceinline__ void async_wait_all() {
#if defined(__gfx1250__)
#if __has_builtin(__builtin_amdgcn_s_wait_asynccnt)
    __builtin_amdgcn_s_wait_asynccnt(0);
#else
    asm volatile("s_wait_asynccnt 0x0" ::: "memory");
#endif
#endif
}

// ---------------------------------------------------------------------------
// fp32 -> bf16 conversion (sizes are multiples of 4)
// ---------------------------------------------------------------------------
__global__ __launch_bounds__(256)
void f32_to_bf16_kernel(const float* __restrict__ in, __bf16* __restrict__ out, size_t n) {
    size_t i = ((size_t)blockIdx.x * 256 + threadIdx.x) * 4;
    if (i + 3 < n) {
        float4 f = *(const float4*)(in + i);
        v4bf o;
        o[0] = (__bf16)f.x; o[1] = (__bf16)f.y; o[2] = (__bf16)f.z; o[3] = (__bf16)f.w;
        *(v4bf*)(out + i) = o;
    }
}

// ---------------------------------------------------------------------------
// C_bf16[M,N] = bf16( A_bf16[M,K] @ W_bf16[N,K]^T + bias[N] ), fp32 accumulate
// block tile 128x128, K-tile 64, 256 threads = 8 waves (2x4 wave grid)
// Double-buffered LDS, tiles staged with async global->LDS (ASYNCcnt).
// ---------------------------------------------------------------------------
__global__ __launch_bounds__(256)
void gemm_bf16_nt(const __bf16* __restrict__ A, const __bf16* __restrict__ W,
                  const float* __restrict__ bias, __bf16* __restrict__ C,
                  int M, int N, int K) {
    __shared__ __attribute__((aligned(16))) __bf16 sA[2][128 * 64];
    __shared__ __attribute__((aligned(16))) __bf16 sB[2][128 * 64];

    const int tid  = threadIdx.x;
    const int lane = tid & 31;
    const int wave = tid >> 5;
    const int wy   = wave >> 1;     // 0..3 : 32-row slab
    const int wx   = wave & 1;      // 0..1 : 64-col slab
    const int row0 = blockIdx.y * 128;
    const int col0 = blockIdx.x * 128;

    v8f acc[2][4];
#pragma unroll
    for (int r = 0; r < 2; r++)
#pragma unroll
        for (int c = 0; c < 4; c++)
            acc[r][c] = v8f{0.f, 0.f, 0.f, 0.f, 0.f, 0.f, 0.f, 0.f};

    const int kbase = (lane >> 4) * 8;          // bf16 A/B fragment K split
    const int arow  = wy * 32 + (lane & 15);    // + r*16
    const int bcol  = wx * 64 + (lane & 15);    // + c*16

    // issue one K-tile's worth of async copies: 128x64 bf16 per matrix,
    // 16B per lane-op, 8 consecutive threads cover a 128-byte row
    auto issue_tile = [&](int buf, int kt) {
#pragma unroll
        for (int t = 0; t < 4; t++) {
            int e  = t * 256 + tid;             // 0..1023
            int r  = e >> 3;                    // 0..127
            int kc = (e & 7) * 8;               // 0..56
            async_copy_b128(&A[(size_t)(row0 + r) * K + kt + kc], &sA[buf][r * 64 + kc]);
            async_copy_b128(&W[(size_t)(col0 + r) * K + kt + kc], &sB[buf][r * 64 + kc]);
        }
    };

    issue_tile(0, 0);

    int cur = 0;
    for (int kt = 0; kt < K; kt += 64, cur ^= 1) {
        async_wait_all();
        __syncthreads();
        if (kt + 64 < K) issue_tile(cur ^ 1, kt + 64);

#pragma unroll
        for (int kk = 0; kk < 64; kk += 32) {
            Frag16 a[2], b[4];
#pragma unroll
            for (int r = 0; r < 2; r++) {
                int rr = arow + r * 16;
                a[r].h[0] = *(const v8bf*)&sA[cur][rr * 64 + kk + kbase];
                a[r].h[1] = *(const v8bf*)&sA[cur][rr * 64 + kk + kbase + 16];
            }
#pragma unroll
            for (int c = 0; c < 4; c++) {
                int cc = bcol + c * 16;
                b[c].h[0] = *(const v8bf*)&sB[cur][cc * 64 + kk + kbase];
                b[c].h[1] = *(const v8bf*)&sB[cur][cc * 64 + kk + kbase + 16];
            }
#pragma unroll
            for (int r = 0; r < 2; r++)
#pragma unroll
                for (int c = 0; c < 4; c++)
                    acc[r][c] = wmma_bf16(a[r], b[c], acc[r][c]);
        }
    }

    // epilogue: C layout -> M = v + 8*(lane>>4), N = lane&15 within each tile
    const int mbase = row0 + wy * 32 + 8 * (lane >> 4);
    const int nbase = col0 + wx * 64 + (lane & 15);
#pragma unroll
    for (int c = 0; c < 4; c++) {
        int n    = nbase + c * 16;
        float bs = bias[n];
#pragma unroll
        for (int r = 0; r < 2; r++) {
#pragma unroll
            for (int v = 0; v < 8; v++) {
                int m = mbase + r * 16 + v;
                C[(size_t)m * N + n] = (__bf16)(acc[r][c][v] + bs);
            }
        }
    }
}

// ---------------------------------------------------------------------------
// logits[B,64] = x_bf16 @ Wr[:, :E]^T + att_bf16 @ Wr[:, E:]^T   (fp32 out)
// block = 128 rows, 8 waves: wave w handles rows [w*16, w*16+16), all 64 cols
// W_router K-tile double-buffered in LDS via async copies; activations
// streamed straight from global (K-contiguous 16B loads).
// ---------------------------------------------------------------------------
__global__ __launch_bounds__(256)
void router_gemm(const __bf16* __restrict__ Xb, const __bf16* __restrict__ Ab,
                 const __bf16* __restrict__ Wr, float* __restrict__ logits,
                 int E) {
    __shared__ __attribute__((aligned(16))) __bf16 sB[2][64 * 64];

    const int tid  = threadIdx.x;
    const int lane = tid & 31;
    const int wave = tid >> 5;
    const int row0 = blockIdx.x * 128;

    v8f acc[4];
#pragma unroll
    for (int c = 0; c < 4; c++)
        acc[c] = v8f{0.f, 0.f, 0.f, 0.f, 0.f, 0.f, 0.f, 0.f};

    const int kbase = (lane >> 4) * 8;
    const int arow  = row0 + wave * 16 + (lane & 15);
    const int twoE  = 2 * E;
    const int steps = twoE / 64;   // flattened over [x | attended] K-space

    auto issue_tile = [&](int buf, int s) {
#pragma unroll
        for (int t = 0; t < 2; t++) {
            int e  = t * 256 + tid;         // 0..511
            int r  = e >> 3;                // 0..63
            int kc = (e & 7) * 8;           // 0..56
            async_copy_b128(&Wr[(size_t)r * twoE + s * 64 + kc], &sB[buf][r * 64 + kc]);
        }
    };

    issue_tile(0, 0);

    int cur = 0;
    for (int s = 0; s < steps; s++, cur ^= 1) {
        async_wait_all();
        __syncthreads();
        if (s + 1 < steps) issue_tile(cur ^ 1, s + 1);

        const int kflat = s * 64;
        const __bf16* Asrc = (kflat < E) ? Xb : Ab;
        const int kt = (kflat < E) ? kflat : (kflat - E);

#pragma unroll
        for (int kk = 0; kk < 64; kk += 32) {
            Frag16 a, b[4];
            a.h[0] = *(const v8bf*)&Asrc[(size_t)arow * E + kt + kk + kbase];
            a.h[1] = *(const v8bf*)&Asrc[(size_t)arow * E + kt + kk + kbase + 16];
#pragma unroll
            for (int c = 0; c < 4; c++) {
                int cc = c * 16 + (lane & 15);
                b[c].h[0] = *(const v8bf*)&sB[cur][cc * 64 + kk + kbase];
                b[c].h[1] = *(const v8bf*)&sB[cur][cc * 64 + kk + kbase + 16];
            }
#pragma unroll
            for (int c = 0; c < 4; c++)
                acc[c] = wmma_bf16(a, b[c], acc[c]);
        }
    }

    const int m0 = row0 + wave * 16 + 8 * (lane >> 4);
#pragma unroll
    for (int c = 0; c < 4; c++) {
        int n = c * 16 + (lane & 15);
#pragma unroll
        for (int v = 0; v < 8; v++)
            logits[(size_t)(m0 + v) * 64 + n] = acc[c][v];
    }
}

// ---------------------------------------------------------------------------
// per-row: top-2 (+softmax of top-2), full softmax -> entropy + expert load
// ---------------------------------------------------------------------------
__global__ __launch_bounds__(256)
void router_finalize(const float* __restrict__ logits, float* __restrict__ out_idx,
                     float* __restrict__ out_w, float* __restrict__ load_acc,
                     float* __restrict__ ent_acc) {
    __shared__ float sLoad[64];
    __shared__ float sEnt;
    const int tid = threadIdx.x;
    if (tid < 64) sLoad[tid] = 0.f;
    if (tid == 0) sEnt = 0.f;
    __syncthreads();

    const int row = blockIdx.x * 256 + tid;
    float l[64];
    const float4* lp = (const float4*)(logits + (size_t)row * 64);
#pragma unroll
    for (int i = 0; i < 16; i++) {
        float4 q = lp[i];
        l[4 * i + 0] = q.x; l[4 * i + 1] = q.y; l[4 * i + 2] = q.z; l[4 * i + 3] = q.w;
    }

    // top-2
    float m1 = -INFINITY, m2 = -INFINITY;
    int   i1 = 0, i2 = 0;
#pragma unroll
    for (int i = 0; i < 64; i++) {
        float v = l[i];
        if (v > m1)      { m2 = m1; i2 = i1; m1 = v; i1 = i; }
        else if (v > m2) { m2 = v;  i2 = i; }
    }

    // full softmax statistics
    float sum = 0.f;
#pragma unroll
    for (int i = 0; i < 64; i++) sum += __expf(l[i] - m1);
    const float inv = 1.f / sum;
    float ent = 0.f;
#pragma unroll
    for (int i = 0; i < 64; i++) {
        float pi = __expf(l[i] - m1) * inv;
        ent -= pi * __logf(pi + 1e-8f);
        atomicAdd(&sLoad[i], pi);
    }
    atomicAdd(&sEnt, ent);

    // softmax over the two top logits
    const float e2 = __expf(m2 - m1);
    const float w1 = 1.f / (1.f + e2);
    out_idx[row * 2 + 0] = (float)i1;
    out_idx[row * 2 + 1] = (float)i2;
    out_w[row * 2 + 0]   = w1;
    out_w[row * 2 + 1]   = e2 * w1;

    __syncthreads();
    if (tid < 64) atomicAdd(&load_acc[tid], sLoad[tid]);
    if (tid == 0) atomicAdd(ent_acc, sEnt);
}

// ---------------------------------------------------------------------------
// scalars: unbiased variance of expert load, mean entropy
// ---------------------------------------------------------------------------
__global__ void finalize_stats(const float* __restrict__ load_acc,
                               const float* __restrict__ ent_acc,
                               float* __restrict__ out_var,
                               float* __restrict__ out_ent, int B) {
    if (threadIdx.x == 0 && blockIdx.x == 0) {
        float el[64];
        float mean = 0.f;
        for (int i = 0; i < 64; i++) { el[i] = load_acc[i] / (float)B; mean += el[i]; }
        mean /= 64.f;
        float var = 0.f;
        for (int i = 0; i < 64; i++) { float d = el[i] - mean; var += d * d; }
        *out_var = var / 63.f;
        *out_ent = ent_acc[0] / (float)B;
    }
}

// ---------------------------------------------------------------------------
extern "C" void kernel_launch(void* const* d_in, const int* in_sizes, int n_in,
                              void* d_out, int out_size, void* d_ws, size_t ws_size,
                              hipStream_t stream) {
    const float* x  = (const float*)d_in[0];   // [B,E]
    const float* Wv = (const float*)d_in[1];   // [E,E]
    const float* bv = (const float*)d_in[2];   // [E]
    const float* Wo = (const float*)d_in[3];   // [E,E]
    const float* bo = (const float*)d_in[4];   // [E]
    const float* Wr = (const float*)d_in[5];   // [N, 2E]
    // d_in[6..11] (context encoder) are dead code in the reference outputs; d_in[12] = top_k (2)

    constexpr int B = 8192, E = 4096, N = 64;

    char*  ws  = (char*)d_ws;
    size_t off = 0;
    auto alloc = [&](size_t bytes) -> void* {
        void* p = ws + off;
        off += (bytes + 255) & ~(size_t)255;
        return p;
    };
    __bf16* xb   = (__bf16*)alloc((size_t)B * E * 2);       // x in bf16
    __bf16* wvb  = (__bf16*)alloc((size_t)E * E * 2);       // Wv in bf16
    __bf16* wob  = (__bf16*)alloc((size_t)E * E * 2);       // Wo in bf16
    __bf16* wrb  = (__bf16*)alloc((size_t)N * 2 * E * 2);   // W_router in bf16
    __bf16* vb   = (__bf16*)alloc((size_t)B * E * 2);       // V = x@Wv.T+bv
    __bf16* ab   = (__bf16*)alloc((size_t)B * E * 2);       // attended
    float*  load_acc = (float*)alloc(64 * sizeof(float));
    float*  ent_acc  = (float*)alloc(sizeof(float));

    float* out        = (float*)d_out;
    float* out_logits = out;                                  // [B,64]
    float* out_idx    = out_logits + (size_t)B * N;           // [B,2] as float
    float* out_w      = out_idx + (size_t)B * 2;              // [B,2]
    float* out_var    = out_w + (size_t)B * 2;                // scalar
    float* out_ent    = out_var + 1;                          // scalar

    (void)hipMemsetAsync(load_acc, 0, 512, stream);  // zeroes load_acc + ent_acc (padded alloc)

    // fp32 -> bf16 conversions
    auto cvt = [&](const float* src, __bf16* dst, size_t n) {
        f32_to_bf16_kernel<<<(unsigned)((n / 4 + 255) / 256), 256, 0, stream>>>(src, dst, n);
    };
    cvt(x,  xb,  (size_t)B * E);
    cvt(Wv, wvb, (size_t)E * E);
    cvt(Wo, wob, (size_t)E * E);
    cvt(Wr, wrb, (size_t)N * 2 * E);

    dim3 gGemm(E / 128, B / 128);  // (32, 64)
    gemm_bf16_nt<<<gGemm, 256, 0, stream>>>(xb, wvb, bv, vb, B, E, E);  // V
    gemm_bf16_nt<<<gGemm, 256, 0, stream>>>(vb, wob, bo, ab, B, E, E);  // attended

    router_gemm<<<B / 128, 256, 0, stream>>>(xb, ab, wrb, out_logits, E);

    router_finalize<<<B / 256, 256, 0, stream>>>(out_logits, out_idx, out_w,
                                                 load_acc, ent_acc);
    finalize_stats<<<1, 64, 0, stream>>>(load_acc, ent_acc, out_var, out_ent, B);
}
